// MultiHeadDotProductAttention_28011776704811
// MI455X (gfx1250) — matrix operand order repaired and checked by reading, not compiled, not run on hardware
//
#include <hip/hip_runtime.h>
#include <hip/hip_bf16.h>

// Problem constants (from reference)
#define E_   4
#define B_   4
#define N_   1024
#define FIN  256
#define FOUT 512
#define NH_  8
#define DK   64

constexpr int M_TOT = B_ * N_;   // 4096 flattened (b,n) rows
constexpr int BLK_M = 128;
constexpr int BLK_N = 64;
constexpr int BLK_K = 32;

typedef __attribute__((ext_vector_type(16))) _Float16 v16h;
typedef __attribute__((ext_vector_type(8)))  float    v8f;

// ---------------------------------------------------------------------------
// Kernel 1: Q/K projection GEMM.  qk[g] (g = e*2 + {0:q,1:k}) = f16(h @ W[e] + b[e])
// 8 waves / block, block tile 128x64, wave tile 32x32 (2x2 WMMA frags), K-step 32.
// ---------------------------------------------------------------------------
__global__ __launch_bounds__(256) void proj_kernel(
    const float* __restrict__ hsrc,
    const float* __restrict__ Wq, const float* __restrict__ bq,
    const float* __restrict__ Wk, const float* __restrict__ bk,
    _Float16* __restrict__ qk)
{
    __shared__ _Float16 As [BLK_M][BLK_K + 8];   // row-major (m,k)
    __shared__ _Float16 BsT[BLK_N][BLK_K + 8];   // n-major, k contiguous

    const int t    = threadIdx.x;
    const int lane = t & 31;
    const int wave = t >> 5;
    const int m0   = blockIdx.x * BLK_M;
    const int f0   = blockIdx.y * BLK_N;
    const int g    = blockIdx.z;            // e*2 + (0=q, 1=k)
    const int e    = g >> 1;
    const bool isK = g & 1;

    const float* W    = (isK ? Wk : Wq) + (size_t)e * FIN * FOUT;
    const float* bias = (isK ? bk : bq) + (size_t)e * FOUT;
    _Float16*    dst  = qk + (size_t)g * M_TOT * FOUT;

    const int wm0    = (wave >> 1) * 32;    // 0,32,64,96
    const int wn0    = (wave & 1) * 32;     // 0,32
    const int laneHi = lane >> 4;           // lane-half (0: lanes 0-15, 1: 16-31)
    const int lr     = lane & 15;

    v8f acc[2][2] = {};

    for (int k0 = 0; k0 < FIN; k0 += BLK_K) {
        __syncthreads();
        // Stage A tile 128x32 (f32 -> f16)
        #pragma unroll
        for (int i = 0; i < (BLK_M * BLK_K) / 256; ++i) {
            int idx = t + i * 256;
            int r = idx >> 5, c = idx & 31;
            As[r][c] = (_Float16)hsrc[(size_t)(m0 + r) * FIN + k0 + c];
        }
        // Stage B tile 32x64, transposed so K is contiguous per output column
        #pragma unroll
        for (int i = 0; i < (BLK_K * BLK_N) / 256; ++i) {
            int idx = t + i * 256;
            int r = idx >> 6, c = idx & 63;        // r = k row, c = n col
            BsT[c][r] = (_Float16)W[(size_t)(k0 + r) * FOUT + f0 + c];
        }
        __syncthreads();

        // A fragment (16x32) in this code's contiguous k order: lanes 0-15 hold K 0-15,
        // lanes 16-31 hold K 16-31, row M = lane & 15.
        v16h afrag[2], bfrag[2];
        #pragma unroll
        for (int ti = 0; ti < 2; ++ti) {
            const _Float16* ap = &As[wm0 + ti * 16 + lr][laneHi * 16];
            #pragma unroll
            for (int j = 0; j < 8; ++j) afrag[ti][j]     = ap[j];
            #pragma unroll
            for (int j = 0; j < 8; ++j) afrag[ti][8 + j] = ap[8 + j];
        }
        // B fragment (32x16): lanes 0-15 hold K 0-15, lanes 16-31 hold K 16-31,
        // column N = lane & 15; K contiguous in BsT row.
        #pragma unroll
        for (int tj = 0; tj < 2; ++tj) {
            const _Float16* bp = &BsT[wn0 + tj * 16 + lr][laneHi * 16];
            #pragma unroll
            for (int j = 0; j < 16; ++j) bfrag[tj][j] = bp[j];
        }
        #pragma unroll
        for (int ti = 0; ti < 2; ++ti)
            #pragma unroll
            for (int tj = 0; tj < 2; ++tj)
                acc[ti][tj] = __builtin_amdgcn_wmma_f32_16x16x32_f16(
                    false, afrag[ti], false, bfrag[tj],
                    (short)0, acc[ti][tj], false, false);
    }

    // Epilogue: C layout — VGPR r: lanes 0-15 -> M=r, lanes 16-31 -> M=r+8; N = lane & 15
    #pragma unroll
    for (int ti = 0; ti < 2; ++ti)
        #pragma unroll
        for (int tj = 0; tj < 2; ++tj)
            #pragma unroll
            for (int r = 0; r < 8; ++r) {
                int row = m0 + wm0 + ti * 16 + (laneHi ? r + 8 : r);
                int col = f0 + wn0 + tj * 16 + lr;
                float v = acc[ti][tj][r] + bias[col];
                dst[(size_t)row * FOUT + col] = (_Float16)v;
            }
}

// ---------------------------------------------------------------------------
// Kernel 2: scores.  One block per (b, n-tile16, m-tile16); wave = head.
// raw_e = Q_e . K_e^T over d_k=64 (2 WMMAs of K=32 per edge type),
// select by adj, zero -> -1e9, write f32 in (h, b, n, m) layout.
// ---------------------------------------------------------------------------
__global__ __launch_bounds__(256) void scores_kernel(
    const _Float16* __restrict__ qk,
    const int* __restrict__ adj,
    float* __restrict__ out)
{
    const int m0   = blockIdx.x * 16;
    const int n0   = blockIdx.y * 16;
    const int b    = blockIdx.z;
    const int t    = threadIdx.x;
    const int lane = t & 31;
    const int head = t >> 5;                 // wave id == head (8 waves)
    const int laneHi = lane >> 4;
    const int lr     = lane & 15;

    const float scale = 0.125f;              // 1/sqrt(64)

    v8f acc[E_] = {};
    #pragma unroll
    for (int e = 0; e < E_; ++e) {
        const _Float16* Qp = qk + ((size_t)(e * 2 + 0) * M_TOT + (size_t)b * N_) * FOUT;
        const _Float16* Kp = qk + ((size_t)(e * 2 + 1) * M_TOT + (size_t)b * N_) * FOUT;
        #pragma unroll
        for (int kk = 0; kk < 2; ++kk) {
            const int dbase = head * DK + kk * 32;
            v16h afrag;
            const _Float16* ap = Qp + (size_t)(n0 + lr) * FOUT + dbase + laneHi * 16;
            #pragma unroll
            for (int j = 0; j < 8; ++j) afrag[j]     = ap[j];
            #pragma unroll
            for (int j = 0; j < 8; ++j) afrag[8 + j] = ap[8 + j];
            // B fragment: K (=d) contiguous in memory -> single 32B vector load
            const _Float16* bp = Kp + (size_t)(m0 + lr) * FOUT + dbase + laneHi * 16;
            v16h bfrag = *(const v16h*)bp;
            acc[e] = __builtin_amdgcn_wmma_f32_16x16x32_f16(
                false, afrag, false, bfrag, (short)0, acc[e], false, false);
        }
    }

    #pragma unroll
    for (int r = 0; r < 8; ++r) {
        const int n = n0 + (laneHi ? r + 8 : r);
        const int m = m0 + lr;
        const int av = adj[((size_t)b * N_ + n) * N_ + m];
        float s = 0.0f;
        if      (av == 1) s = acc[0][r] * scale;
        else if (av == 2) s = acc[1][r] * scale;
        else if (av == 3) s = acc[2][r] * scale;
        else if (av == 4) s = acc[3][r] * scale;
        if (s == 0.0f) s = -1.0e9f;
        out[(((size_t)head * B_ + b) * N_ + n) * N_ + m] = s;
    }
}

// ---------------------------------------------------------------------------
// Kernel 3: in-place row softmax over the last axis (rows of 1024 floats).
// One 256-thread block (8 wave32) per row.
// ---------------------------------------------------------------------------
__global__ __launch_bounds__(256) void softmax_kernel(float* __restrict__ out)
{
    float* p = out + (size_t)blockIdx.x * N_;
    const int t    = threadIdx.x;
    const int lane = t & 31;
    const int wave = t >> 5;
    __shared__ float red[8];

    float x[4];
    float mx = -3.4e38f;
    #pragma unroll
    for (int i = 0; i < 4; ++i) { x[i] = p[t + i * 256]; mx = fmaxf(mx, x[i]); }
    #pragma unroll
    for (int off = 16; off > 0; off >>= 1) mx = fmaxf(mx, __shfl_xor(mx, off, 32));
    if (lane == 0) red[wave] = mx;
    __syncthreads();
    mx = red[0];
    #pragma unroll
    for (int w = 1; w < 8; ++w) mx = fmaxf(mx, red[w]);
    __syncthreads();

    float sum = 0.0f;
    #pragma unroll
    for (int i = 0; i < 4; ++i) { x[i] = __expf(x[i] - mx); sum += x[i]; }
    #pragma unroll
    for (int off = 16; off > 0; off >>= 1) sum += __shfl_xor(sum, off, 32);
    if (lane == 0) red[wave] = sum;
    __syncthreads();
    sum = 0.0f;
    #pragma unroll
    for (int w = 0; w < 8; ++w) sum += red[w];
    const float inv = 1.0f / sum;
    #pragma unroll
    for (int i = 0; i < 4; ++i) p[t + i * 256] = x[i] * inv;
}

// ---------------------------------------------------------------------------
extern "C" void kernel_launch(void* const* d_in, const int* in_sizes, int n_in,
                              void* d_out, int out_size, void* d_ws, size_t ws_size,
                              hipStream_t stream)
{
    const float* hsrc = (const float*)d_in[0];
    const int*   adj  = (const int*)  d_in[1];
    const float* Wq   = (const float*)d_in[2];
    const float* bq   = (const float*)d_in[3];
    const float* Wk   = (const float*)d_in[4];
    const float* bk   = (const float*)d_in[5];
    float*       out  = (float*)d_out;
    _Float16*    qk   = (_Float16*)d_ws;   // 8 * 4096 * 512 f16 = 32 MB

    proj_kernel<<<dim3(M_TOT / BLK_M, FOUT / BLK_N, 2 * E_), 256, 0, stream>>>(
        hsrc, Wq, bq, Wk, bk, qk);
    scores_kernel<<<dim3(N_ / 16, N_ / 16, B_), 256, 0, stream>>>(qk, adj, out);
    softmax_kernel<<<NH_ * B_ * N_, 256, 0, stream>>>(out);
}
